// CodecLlamaCodecEmbedding_56461640073704
// MI455X (gfx1250) — matrix operand order repaired and checked by reading, not compiled, not run on hardware
//
#include <hip/hip_runtime.h>
#include <hip/hip_bf16.h>
#include <math.h>

// ---------------- problem constants ----------------
#define NUM_CODEBOOKS 8
#define CODEBOOK_SIZE 2048
#define CODEBOOK_DIM  16
#define HIDDEN        2048
#define NTOK          16384           // B*S = 4*4096
#define PAD_SLOTS     17408           // 136 * 128  (<= NTOK + 8*127, rounded)
#define NSUPER        136             // 128-row supertiles

// ---------------- workspace layout (bytes) ----------------
#define WS_CNT   0                         // 8 ints
#define WS_OFFS  32                        // 8 ints
#define WS_BASE  64                        // 8 ints
#define WS_TMAP  96                        // 136 ints
#define WS_PERM  1024                      // 17408 ints  (69632 B)
#define WS_W2T   131072                    // 8*2048*2048*2 = 67108864 B (bf16)
#define WS_HBUF  (WS_W2T + 67108864)       // 17408*2048*2 = 71303168 B (bf16)

// ---------------- types for WMMA ----------------
typedef __attribute__((ext_vector_type(16))) __bf16 v16bf;
typedef __attribute__((ext_vector_type(8)))  float  v8f;

union BFrag {
    v16bf v;
    uint4 u[2];
};

static __device__ __forceinline__ unsigned short f2bf(float x) {
    unsigned u = __builtin_bit_cast(unsigned, x);
    u += 0x7FFFu + ((u >> 16) & 1u);     // round-to-nearest-even
    return (unsigned short)(u >> 16);
}

static __device__ __forceinline__ float gelu_exact(float x) {
    return 0.5f * x * (1.0f + erff(x * 0.70710678118654752f));
}

// ============ 1) init: zero counters, fill perm with -1 ============
__global__ void k_init(int* cnt, int* offs, int* perm) {
    int idx = blockIdx.x * 256 + threadIdx.x;
    if (blockIdx.x == 0 && threadIdx.x < 8) { cnt[threadIdx.x] = 0; offs[threadIdx.x] = 0; }
    if (idx < PAD_SLOTS) perm[idx] = -1;
}

// ============ 2) count tokens per codebook ============
__global__ void k_count(const int* __restrict__ ids, int* cnt) {
    int t = blockIdx.x * 256 + threadIdx.x;
    if (t < NTOK) {
        int cb = ids[t] >> 11;             // id / CODEBOOK_SIZE
        atomicAdd(&cnt[cb], 1);
    }
}

// ============ 3) serial scan: 128-aligned bucket bases + tile map ============
__global__ void k_scan(const int* __restrict__ cnt, int* base, int* tmap) {
    if (threadIdx.x != 0 || blockIdx.x != 0) return;
    for (int t = 0; t < NSUPER; ++t) tmap[t] = -1;
    int cur = 0;
    for (int k = 0; k < NUM_CODEBOOKS; ++k) {
        base[k] = cur;
        int end = cur + cnt[k];
        int t0 = cur >> 7;
        int t1 = (end + 127) >> 7;
        for (int t = t0; t < t1; ++t) tmap[t] = k;
        cur = (end + 127) & ~127;
    }
}

// ============ 4) scatter tokens into bucket-ordered perm ============
__global__ void k_scatter(const int* __restrict__ ids, const int* __restrict__ base,
                          int* offs, int* perm) {
    int t = blockIdx.x * 256 + threadIdx.x;
    if (t < NTOK) {
        int cb = ids[t] >> 11;
        int pos = base[cb] + atomicAdd(&offs[cb], 1);
        perm[pos] = t;
    }
}

// ============ 5) W2 fp32 -> bf16, transposed:  W2t[k][g][h] ============
__global__ void k_convert_w2(const float* __restrict__ W2, unsigned short* __restrict__ w2t) {
    __shared__ float tile[32][33];
    int kb = blockIdx.z;
    int h0 = blockIdx.y * 32;
    int g0 = blockIdx.x * 32;
    int tx = threadIdx.x & 31;
    int ty = threadIdx.x >> 5;             // 0..7
    const float* src = W2 + ((size_t)kb * HIDDEN + h0) * HIDDEN + g0;
#pragma unroll
    for (int i = 0; i < 4; ++i)
        tile[ty + i * 8][tx] = src[(size_t)(ty + i * 8) * HIDDEN + tx];
    __syncthreads();
    unsigned short* dst = w2t + ((size_t)kb * HIDDEN + g0) * HIDDEN + h0;
#pragma unroll
    for (int i = 0; i < 4; ++i)
        dst[(size_t)(ty + i * 8) * HIDDEN + tx] = f2bf(tile[tx][ty + i * 8]);
}

// ============ 6) GEMM1 (16->2048) + exact GELU, bf16 h in bucket order ============
// one block per slot; 256 threads x 8 outputs = 2048
__global__ void k_gemm1_gelu(const int* __restrict__ perm, const int* __restrict__ ids,
                             const float* __restrict__ emb, const float* __restrict__ W1,
                             const float* __restrict__ b1, unsigned short* __restrict__ hbuf) {
    int s = blockIdx.x;
    int j0 = threadIdx.x * 8;
    unsigned short* op = hbuf + (size_t)s * HIDDEN + j0;
    int t = perm[s];
    if (t < 0) {
        uint4 z = make_uint4(0u, 0u, 0u, 0u);
        *(uint4*)op = z;
        return;
    }
    int id = ids[t];
    int k = id >> 11;
    const float4* ep = (const float4*)(emb + (size_t)id * CODEBOOK_DIM);
    float e[16];
    float4 e4;
    e4 = ep[0]; e[0]=e4.x; e[1]=e4.y; e[2]=e4.z; e[3]=e4.w;
    e4 = ep[1]; e[4]=e4.x; e[5]=e4.y; e[6]=e4.z; e[7]=e4.w;
    e4 = ep[2]; e[8]=e4.x; e[9]=e4.y; e[10]=e4.z; e[11]=e4.w;
    e4 = ep[3]; e[12]=e4.x; e[13]=e4.y; e[14]=e4.z; e[15]=e4.w;

    const float* w = W1 + (size_t)k * CODEBOOK_DIM * HIDDEN + j0;
    const float* bp = b1 + (size_t)k * HIDDEN + j0;
    float acc[8];
    {
        float4 bA = *(const float4*)(bp);
        float4 bB = *(const float4*)(bp + 4);
        acc[0]=bA.x; acc[1]=bA.y; acc[2]=bA.z; acc[3]=bA.w;
        acc[4]=bB.x; acc[5]=bB.y; acc[6]=bB.z; acc[7]=bB.w;
    }
#pragma unroll
    for (int d = 0; d < CODEBOOK_DIM; ++d) {
        float ev = e[d];
        float4 w0 = *(const float4*)(w + (size_t)d * HIDDEN);
        float4 w1v = *(const float4*)(w + (size_t)d * HIDDEN + 4);
        acc[0] = fmaf(ev, w0.x, acc[0]);
        acc[1] = fmaf(ev, w0.y, acc[1]);
        acc[2] = fmaf(ev, w0.z, acc[2]);
        acc[3] = fmaf(ev, w0.w, acc[3]);
        acc[4] = fmaf(ev, w1v.x, acc[4]);
        acc[5] = fmaf(ev, w1v.y, acc[5]);
        acc[6] = fmaf(ev, w1v.z, acc[6]);
        acc[7] = fmaf(ev, w1v.w, acc[7]);
    }
    unsigned short h16[8];
#pragma unroll
    for (int j = 0; j < 8; ++j) h16[j] = f2bf(gelu_exact(acc[j]));
    *(uint4*)op = *(const uint4*)h16;
}

// ============ 7) GEMM2: bf16 WMMA 128x128 tiles, scatter epilogue ============
// 8 waves: wave grid 2(M)x4(N); each wave 64x32 = 4x2 accumulators.
// Only two 64-bit base addresses live in the K loop; the per-fragment row
// deltas (mi*16*HIDDEN, ni*16*HIDDEN elements) fold into the 24-bit signed
// instruction offset of global_load_b128, eliminating the address spills.
__global__ void k_gemm2_wmma(const int* __restrict__ perm, const int* __restrict__ tmap,
                             const unsigned short* __restrict__ hbuf,
                             const unsigned short* __restrict__ w2t,
                             const float* __restrict__ b2, float* __restrict__ out) {
    int st = blockIdx.y;
    int k = tmap[st];
    if (k < 0) return;                       // empty supertile

    int lane = threadIdx.x & 31;
    int wid  = threadIdx.x >> 5;
    int wm = wid & 1;                        // 2 M groups
    int wn = wid >> 1;                       // 4 N groups
    int row0 = st * 128 + wm * 64;
    int col0 = blockIdx.x * 128 + wn * 32;
    int lr = lane & 15;
    int lh = lane >> 4;

    v8f zero = {0.f, 0.f, 0.f, 0.f, 0.f, 0.f, 0.f, 0.f};
    v8f acc[4][2];
#pragma unroll
    for (int mi = 0; mi < 4; ++mi)
#pragma unroll
        for (int ni = 0; ni < 2; ++ni) acc[mi][ni] = zero;

    // A fragment per lane: row (row0 + mi*16 + lr); lane half 0 -> K {0..7,16..23},
// half 1 -> K {8..15,24..31}.  B fragment per lane: W2t row (col0 + ni*16 + lr);
    // same K split as A: half 0 -> K {0..7,16..23}, half 1 -> K {8..15,24..31}.
    const unsigned short* ap = hbuf + (size_t)(row0 + lr) * HIDDEN + lh * 8;
    const unsigned short* bp =
        w2t + (size_t)k * HIDDEN * HIDDEN + (size_t)(col0 + lr) * HIDDEN + lh * 8;

    for (int kk0 = 0; kk0 < HIDDEN; kk0 += 128) {
        __builtin_prefetch(ap + 128, 0, 3);   // -> global_prefetch_b8, one K-superstep ahead
        __builtin_prefetch(bp + 128, 0, 3);
#pragma unroll
        for (int kk = 0; kk < 128; kk += 32) {
            BFrag a[4], b[2];
#pragma unroll
            for (int mi = 0; mi < 4; ++mi) {
                a[mi].u[0] = *(const uint4*)(ap + kk + mi * 16 * HIDDEN);
                a[mi].u[1] = *(const uint4*)(ap + kk + mi * 16 * HIDDEN + 16);
            }
#pragma unroll
            for (int ni = 0; ni < 2; ++ni) {
                b[ni].u[0] = *(const uint4*)(bp + kk + ni * 16 * HIDDEN);
                b[ni].u[1] = *(const uint4*)(bp + kk + ni * 16 * HIDDEN + 16);
            }
#pragma unroll
            for (int mi = 0; mi < 4; ++mi)
#pragma unroll
                for (int ni = 0; ni < 2; ++ni)
                    acc[mi][ni] = __builtin_amdgcn_wmma_f32_16x16x32_bf16(
                        false, a[mi].v, false, b[ni].v, (short)0, acc[mi][ni],
                        false, false);
        }
        ap += 128;
        bp += 128;
    }

    // epilogue: C layout lane half h -> rows M = h*8 + i, N = lane&15
#pragma unroll
    for (int ni = 0; ni < 2; ++ni) {
        int col = col0 + ni * 16 + lr;
        float bias = b2[(size_t)k * HIDDEN + col];
#pragma unroll
        for (int mi = 0; mi < 4; ++mi) {
#pragma unroll
            for (int i = 0; i < 8; ++i) {
                int s = row0 + mi * 16 + lh * 8 + i;
                int t = perm[s];
                if (t >= 0)
                    out[(size_t)t * HIDDEN + col] = acc[mi][ni][i] + bias;
            }
        }
    }
}

// ============================ launcher ============================
extern "C" void kernel_launch(void* const* d_in, const int* in_sizes, int n_in,
                              void* d_out, int out_size, void* d_ws, size_t ws_size,
                              hipStream_t stream) {
    const int*   ids = (const int*)d_in[0];
    const float* emb = (const float*)d_in[1];
    const float* W1  = (const float*)d_in[2];
    const float* b1  = (const float*)d_in[3];
    const float* W2  = (const float*)d_in[4];
    const float* b2  = (const float*)d_in[5];
    float* out = (float*)d_out;

    char* ws = (char*)d_ws;
    int* cnt   = (int*)(ws + WS_CNT);
    int* offs  = (int*)(ws + WS_OFFS);
    int* base  = (int*)(ws + WS_BASE);
    int* tmap  = (int*)(ws + WS_TMAP);
    int* perm  = (int*)(ws + WS_PERM);
    unsigned short* w2t  = (unsigned short*)(ws + WS_W2T);
    unsigned short* hbuf = (unsigned short*)(ws + WS_HBUF);

    k_init<<<dim3((PAD_SLOTS + 255) / 256), dim3(256), 0, stream>>>(cnt, offs, perm);
    k_count<<<dim3(NTOK / 256), dim3(256), 0, stream>>>(ids, cnt);
    k_scan<<<dim3(1), dim3(32), 0, stream>>>(cnt, base, tmap);
    k_scatter<<<dim3(NTOK / 256), dim3(256), 0, stream>>>(ids, base, offs, perm);
    k_convert_w2<<<dim3(HIDDEN / 32, HIDDEN / 32, NUM_CODEBOOKS), dim3(256), 0, stream>>>(W2, w2t);
    k_gemm1_gelu<<<dim3(PAD_SLOTS), dim3(256), 0, stream>>>(perm, ids, emb, W1, b1, hbuf);
    k_gemm2_wmma<<<dim3(HIDDEN / 128, NSUPER), dim3(256), 0, stream>>>(perm, tmap, hbuf, w2t, b2, out);
}
